// NeighborTypeAwareGraphAttention_36636071035263
// MI455X (gfx1250) — matrix operand order repaired and checked by reading, not compiled, not run on hardware
//
#include <hip/hip_runtime.h>
#include <hip/hip_bf16.h>
#include <math.h>
#include <stdint.h>

// ---------------------------------------------------------------------------
// NeighborTypeAwareGraphAttention for MI455X (gfx1250, wave32, WMMA)
// N=100000 nodes, E=400000 edges, IN=128, H=8 heads, HF=32, out 256 / 32
// ---------------------------------------------------------------------------

#define IN_DIM 128
#define HID    256   // H*HF
#define HEADS  8
#define HF_DIM 32
#define OUT_DIM 32
#define BROW   136   // LDS row stride (halfs) for staged W: 128 + 8 pad (bank spread)

typedef __attribute__((ext_vector_type(16))) _Float16 v16h;
typedef __attribute__((ext_vector_type(8)))  _Float16 v8h;
typedef __attribute__((ext_vector_type(8)))  float    v8f;

// order-preserving float <-> uint key (for atomic max / radix select)
__device__ __forceinline__ unsigned f2key(float f) {
  unsigned u = __float_as_uint(f);
  return (u & 0x80000000u) ? ~u : (u | 0x80000000u);
}
__device__ __forceinline__ float key2f(unsigned k) {
  unsigned u = (k & 0x80000000u) ? (k & 0x7FFFFFFFu) : ~k;
  return __uint_as_float(u);
}

// mask categories: 0=all, 1=neg (p<negThr), 2=pos (p>posThr), 3=unknown
__device__ __forceinline__ bool mask_of(int id, float p, float negThr, float posThr) {
  switch (id) {
    case 0:  return true;
    case 1:  return p < negThr;
    case 2:  return p > posThr;
    default: return (p <= posThr) && (p >= negThr);
  }
}

__device__ __forceinline__ float lrelu(float x) { return x < 0.0f ? 0.2f * x : x; }

// gfx1250 async global->LDS copy (ASYNCcnt-tracked, no VGPR round trip).
// lds_off = byte offset within LDS (low 32 bits of generic shared pointer),
// gaddr   = 64-bit global address.
__device__ __forceinline__ void async_g2lds_b128(unsigned lds_off, unsigned long long gaddr) {
  asm volatile("global_load_async_to_lds_b128 %0, %1, off"
               :: "v"(lds_off), "v"(gaddr) : "memory");
}
__device__ __forceinline__ void wait_asynccnt0() {
  asm volatile("s_wait_asynccnt 0x0" ::: "memory");
}

// ---------------------------------------------------------------------------
// 1) row-normalize feat, emit f16 copy for WMMA A-matrix
// ---------------------------------------------------------------------------
__global__ __launch_bounds__(256) void k_norm_feat(const float* __restrict__ feat,
                                                   _Float16* __restrict__ featf16, int n) {
  int wave = (blockIdx.x * blockDim.x + threadIdx.x) >> 5;
  int lane = threadIdx.x & 31;
  if (wave >= n) return;
  const float* r = feat + (size_t)wave * IN_DIM;
  float ss = 0.0f;
  for (int i = lane; i < IN_DIM; i += 32) { float v = r[i]; ss += v * v; }
  for (int off = 16; off; off >>= 1) ss += __shfl_xor(ss, off, 32);
  float inv = 1.0f / fmaxf(sqrtf(ss), 1e-12f);
  _Float16* o = featf16 + (size_t)wave * IN_DIM;
  for (int i = lane; i < IN_DIM; i += 32) o[i] = (_Float16)(r[i] * inv);
}

// ---------------------------------------------------------------------------
// 2) masked quantile (torch.quantile linear interpolation) via radix select
//    single workgroup of 1024 threads
// ---------------------------------------------------------------------------
__device__ unsigned block_reduce_add_u(unsigned* red, int tid, unsigned v) {
  red[tid] = v; __syncthreads();
  for (int s = 512; s; s >>= 1) { if (tid < s) red[tid] += red[tid + s]; __syncthreads(); }
  unsigned r = red[0]; __syncthreads();
  return r;
}

__global__ __launch_bounds__(1024) void k_quantile(const float* __restrict__ ep, int e,
                                                   int maskSel, float q,
                                                   float* __restrict__ thrOut) {
  __shared__ unsigned red[1024];
  int tid = threadIdx.x;

  // k = |mask|
  unsigned c = 0;
  for (int i = tid; i < e; i += 1024) {
    float x = ep[i];
    bool m = maskSel ? (x > 0.0f) : (x <= 0.0f);
    if (m) c++;
  }
  unsigned k = block_reduce_add_u(red, tid, c);

  float pos = fmaxf(q * (float)((int)k - 1), 0.0f);
  int lo = (int)floorf(pos);
  int hi = (int)ceilf(pos);
  float frac = pos - (float)lo;

  float vals[2];
  int idxs[2] = {lo, hi};
  for (int t = 0; t < 2; t++) {
    unsigned ans = 0;
    for (int b = 31; b >= 0; b--) {
      unsigned test = ans | (1u << b);
      unsigned cnt = 0;
      for (int i = tid; i < e; i += 1024) {
        float x = ep[i];
        bool m = maskSel ? (x > 0.0f) : (x <= 0.0f);
        if (m && f2key(x) < test) cnt++;
      }
      unsigned total = block_reduce_add_u(red, tid, cnt);
      if (total <= (unsigned)idxs[t]) ans = test;  // keep bit: still <= idx-th value
    }
    vals[t] = key2f(ans);
  }
  if (tid == 0)
    thrOut[0] = (k > 0) ? (vals[0] * (1.0f - frac) + vals[1] * frac) : 0.0f;
}

// ---------------------------------------------------------------------------
// 3) softmax(iw) -> combination scales {1, w0, w1, w2} for convs {g,n,p,u}
// ---------------------------------------------------------------------------
__global__ void k_scales(const float* __restrict__ iw, float* __restrict__ scales) {
  float a = iw[0], b = iw[1], c = iw[2];
  float m = fmaxf(a, fmaxf(b, c));
  float ea = expf(a - m), eb = expf(b - m), ec = expf(c - m);
  float s = ea + eb + ec;
  scales[0] = 1.0f;
  scales[1] = ea / s;
  scales[2] = eb / s;
  scales[3] = ec / s;
}

// ---------------------------------------------------------------------------
// 4) the four masked 1-D softmaxes over -edge_pred -> w_all/w_neg/w_pos/w_unk
//    single workgroup of 1024 threads
// ---------------------------------------------------------------------------
__device__ float block_reduce_max_f(float* red, int tid, float v) {
  red[tid] = v; __syncthreads();
  for (int s = 512; s; s >>= 1) { if (tid < s) red[tid] = fmaxf(red[tid], red[tid + s]); __syncthreads(); }
  float r = red[0]; __syncthreads();
  return r;
}
__device__ float block_reduce_add_f(float* red, int tid, float v) {
  red[tid] = v; __syncthreads();
  for (int s = 512; s; s >>= 1) { if (tid < s) red[tid] += red[tid + s]; __syncthreads(); }
  float r = red[0]; __syncthreads();
  return r;
}

__global__ __launch_bounds__(1024) void k_edge_weights(const float* __restrict__ edge_pred,
                                                       const float* __restrict__ thr,
                                                       float* __restrict__ w, int e) {
  __shared__ float red[1024];
  int tid = threadIdx.x;
  float negThr = thr[0], posThr = thr[1];

  float mx[4] = {-INFINITY, -INFINITY, -INFINITY, -INFINITY};
  for (int i = tid; i < e; i += 1024) {
    float p = edge_pred[i];
    float x = -p;
    mx[0] = fmaxf(mx[0], x);
    if (p < negThr) mx[1] = fmaxf(mx[1], x);
    if (p > posThr) mx[2] = fmaxf(mx[2], x);
    if (p <= posThr && p >= negThr) mx[3] = fmaxf(mx[3], x);
  }
  float M[4];
  for (int c = 0; c < 4; c++) M[c] = block_reduce_max_f(red, tid, mx[c]);

  float sm[4] = {0.f, 0.f, 0.f, 0.f};
  for (int i = tid; i < e; i += 1024) {
    float p = edge_pred[i];
    float x = -p;
    sm[0] += expf(x - M[0]);
    if (p < negThr) sm[1] += expf(x - M[1]);
    if (p > posThr) sm[2] += expf(x - M[2]);
    if (p <= posThr && p >= negThr) sm[3] += expf(x - M[3]);
  }
  float S[4];
  for (int c = 0; c < 4; c++) S[c] = fmaxf(block_reduce_add_f(red, tid, sm[c]), 1e-38f);

  for (int i = tid; i < e; i += 1024) {
    float p = edge_pred[i];
    float x = -p;
    w[0 * e + i] = expf(x - M[0]) / S[0];
    w[1 * e + i] = (p < negThr) ? expf(x - M[1]) / S[1] : 0.0f;
    w[2 * e + i] = (p > posThr) ? expf(x - M[2]) / S[2] : 0.0f;
    w[3 * e + i] = (p <= posThr && p >= negThr) ? expf(x - M[3]) / S[3] : 0.0f;
  }
}

// ---------------------------------------------------------------------------
// 5) combined accumulator init = sum_c scales[c] * b_c[col]
// ---------------------------------------------------------------------------
__global__ __launch_bounds__(256) void k_init_combined(const float* __restrict__ bg,
                                                       const float* __restrict__ bn,
                                                       const float* __restrict__ bp,
                                                       const float* __restrict__ bu,
                                                       const float* __restrict__ scales,
                                                       float* __restrict__ combined, int n) {
  size_t idx = (size_t)blockIdx.x * blockDim.x + threadIdx.x;
  if (idx >= (size_t)n * HID) return;
  int col = (int)(idx & (HID - 1));
  combined[idx] = scales[0] * bg[col] + scales[1] * bn[col] + scales[2] * bp[col] + scales[3] * bu[col];
}

// ---------------------------------------------------------------------------
// 6) f32 -> f16 weight conversion
// ---------------------------------------------------------------------------
__global__ __launch_bounds__(256) void k_cvt_f16(const float* __restrict__ W,
                                                 _Float16* __restrict__ Wh, int count) {
  int i = blockIdx.x * blockDim.x + threadIdx.x;
  if (i < count) Wh[i] = (_Float16)W[i];
}

// ---------------------------------------------------------------------------
// 7) h = featf16 (n x 128) @ W^T  -> (n x 256) f32, WMMA f16 path.
//    One block = one 16-row M-stripe.  W (256x128 f16, 64 KB) is staged into
//    LDS once per block with gfx1250 async global->LDS copies (ASYNCcnt),
//    padded to 136-half rows for bank spread.  Each of the 8 waves computes
//    two 16x16 N-tiles, reusing its A fragment across both WMMAs per K-step.
// ---------------------------------------------------------------------------
__global__ __launch_bounds__(256) void k_gemm_h(const _Float16* __restrict__ A,
                                                const _Float16* __restrict__ B,
                                                float* __restrict__ hout, int n) {
  __shared__ _Float16 Bsh[HID * BROW];  // 256 rows x 136 halfs = 68 KB

  int tid = threadIdx.x;
  // stage W: thread t copies row t (256 B) as 16 async b128 transfers
  {
    unsigned ldsOff = ((unsigned)(uintptr_t)(&Bsh[0])) + (unsigned)tid * (BROW * 2);
    unsigned long long g = (unsigned long long)(uintptr_t)(B + (size_t)tid * IN_DIM);
#pragma unroll
    for (int j = 0; j < 16; j++)
      async_g2lds_b128(ldsOff + j * 16, g + (unsigned long long)j * 16);
    wait_asynccnt0();        // this wave's transfers complete
  }
  __syncthreads();           // all waves' transfers visible

  int wave = tid >> 5;
  int lane = tid & 31;
  int tm = blockIdx.x * 16;
  if (tm >= n) return;
  int tn0 = wave * 32;                         // two N-tiles: tn0, tn0+16
  int l15 = lane & 15;
  int kg  = lane >> 4;                         // 0 or 1

  const _Float16* arow = A + (size_t)(tm + l15) * IN_DIM;
  const _Float16* b0 = &Bsh[(tn0 + l15) * BROW];
  const _Float16* b1 = &Bsh[(tn0 + 16 + l15) * BROW];

  v8f acc0 = {}, acc1 = {};
  for (int k0 = 0; k0 < IN_DIM; k0 += 32) {
    __builtin_prefetch(arow + k0 + 32, 0, 0);  // global_prefetch_b8
    // A frag (contiguous K order, same as B): lane grp 0: K [0..15]; grp 1: [16..31]
    v8h a0 = *(const v8h*)(arow + k0 + kg * 16);
    v8h a1 = *(const v8h*)(arow + k0 + kg * 16 + 8);
    v16h av = __builtin_shufflevector(a0, a1, 0, 1, 2, 3, 4, 5, 6, 7,
                                              8, 9, 10, 11, 12, 13, 14, 15);
    // B frag: lane holds 16 contiguous K of column lane&15, K offset (lane>>4)*16
    v16h bv0 = *(const v16h*)(b0 + k0 + kg * 16);
    v16h bv1 = *(const v16h*)(b1 + k0 + kg * 16);
    acc0 = __builtin_amdgcn_wmma_f32_16x16x32_f16(false, av, false, bv0,
                                                  (short)0, acc0, false, false);
    acc1 = __builtin_amdgcn_wmma_f32_16x16x32_f16(false, av, false, bv1,
                                                  (short)0, acc1, false, false);
  }
  // D layout: VGPR v -> M = v + 8*kg, N = lane&15
  for (int v = 0; v < 8; v++) {
    hout[(size_t)(tm + v + 8 * kg) * HID + tn0 + l15]      = acc0[v];
    hout[(size_t)(tm + v + 8 * kg) * HID + tn0 + 16 + l15] = acc1[v];
  }
}

// ---------------------------------------------------------------------------
// 8) per-(node,head) attention projections el/er
// ---------------------------------------------------------------------------
__global__ __launch_bounds__(256) void k_elr(const float* __restrict__ h,
                                             const float* __restrict__ al,
                                             const float* __restrict__ ar,
                                             float* __restrict__ el,
                                             float* __restrict__ er, int n) {
  int idx = blockIdx.x * blockDim.x + threadIdx.x;
  if (idx >= n * HEADS) return;
  int node = idx >> 3, head = idx & 7;
  const float* hp = h + (size_t)node * HID + head * HF_DIM;
  const float* alp = al + head * HF_DIM;
  const float* arp = ar + head * HF_DIM;
  float sl = 0.f, sr = 0.f;
  for (int f = 0; f < HF_DIM; f++) { float v = hp[f]; sl += v * alp[f]; sr += v * arp[f]; }
  el[idx] = sl;
  er[idx] = sr;
}

// ---------------------------------------------------------------------------
// 9) zero segment max (as key) / segment sum
// ---------------------------------------------------------------------------
__global__ __launch_bounds__(256) void k_zero_seg(unsigned* __restrict__ segmax,
                                                  float* __restrict__ segsum, int cnt) {
  int i = blockIdx.x * blockDim.x + threadIdx.x;
  if (i < cnt) { segmax[i] = 0u; segsum[i] = 0.0f; }
}

// ---------------------------------------------------------------------------
// 10/11) per-(edge,head) segment max then segment sum (masked)
// ---------------------------------------------------------------------------
__global__ __launch_bounds__(256) void k_edge_max(const float* __restrict__ edge_pred,
                                                  const float* __restrict__ thr,
                                                  const int* __restrict__ src,
                                                  const int* __restrict__ dst,
                                                  const float* __restrict__ el,
                                                  const float* __restrict__ er,
                                                  unsigned* __restrict__ segmax,
                                                  int e, int maskId) {
  int idx = blockIdx.x * blockDim.x + threadIdx.x;
  if (idx >= e * HEADS) return;
  int ed = idx >> 3, hd = idx & 7;
  if (!mask_of(maskId, edge_pred[ed], thr[0], thr[1])) return;
  int s = src[ed], d = dst[ed];
  float ev = lrelu(el[s * HEADS + hd] + er[d * HEADS + hd]);
  atomicMax(&segmax[d * HEADS + hd], f2key(ev));
}

__global__ __launch_bounds__(256) void k_edge_sum(const float* __restrict__ edge_pred,
                                                  const float* __restrict__ thr,
                                                  const int* __restrict__ src,
                                                  const int* __restrict__ dst,
                                                  const float* __restrict__ el,
                                                  const float* __restrict__ er,
                                                  const unsigned* __restrict__ segmax,
                                                  float* __restrict__ segsum,
                                                  int e, int maskId) {
  int idx = blockIdx.x * blockDim.x + threadIdx.x;
  if (idx >= e * HEADS) return;
  int ed = idx >> 3, hd = idx & 7;
  if (!mask_of(maskId, edge_pred[ed], thr[0], thr[1])) return;
  int s = src[ed], d = dst[ed];
  float ev = lrelu(el[s * HEADS + hd] + er[d * HEADS + hd]);
  float m = key2f(segmax[d * HEADS + hd]);
  atomicAdd(&segsum[d * HEADS + hd], expf(ev - m));
}

// ---------------------------------------------------------------------------
// 12) weighted scatter: combined[dst] += scale * a * h[src]; one wave / edge
// ---------------------------------------------------------------------------
__global__ __launch_bounds__(256) void k_edge_aggr(const float* __restrict__ edge_pred,
                                                   const float* __restrict__ thr,
                                                   const int* __restrict__ src,
                                                   const int* __restrict__ dst,
                                                   const float* __restrict__ el,
                                                   const float* __restrict__ er,
                                                   const unsigned* __restrict__ segmax,
                                                   const float* __restrict__ segsum,
                                                   const float* __restrict__ w,
                                                   const float* __restrict__ scales,
                                                   int scaleIdx,
                                                   const float* __restrict__ h,
                                                   float* __restrict__ combined,
                                                   int e, int maskId) {
  int wave = (blockIdx.x * blockDim.x + threadIdx.x) >> 5;
  int lane = threadIdx.x & 31;
  if (wave >= e) return;
  float p = edge_pred[wave];
  if (!mask_of(maskId, p, thr[0], thr[1])) return;
  int s = src[wave], d = dst[wave];
  float wsc = w[wave] * scales[scaleIdx];
  const float* hs = h + (size_t)s * HID;
  float* cb = combined + (size_t)d * HID;
  for (int hd = 0; hd < HEADS; hd++) {
    float ev = lrelu(el[s * HEADS + hd] + er[d * HEADS + hd]);
    float m = key2f(segmax[d * HEADS + hd]);
    float a = expf(ev - m) / fmaxf(segsum[d * HEADS + hd], 1e-38f) * wsc;
    int col = hd * 32 + lane;
    atomicAdd(&cb[col], a * hs[col]);
  }
}

// ---------------------------------------------------------------------------
// 13) out = relu(combined (n x 256) @ Wo^T + bo) -> (n x 32), WMMA f16 path,
//     combined converted to f16 on the fly in the A-fragment load.
// ---------------------------------------------------------------------------
__global__ __launch_bounds__(256) void k_gemm_out(const float* __restrict__ combined,
                                                  const _Float16* __restrict__ Woh,
                                                  const float* __restrict__ bo,
                                                  float* __restrict__ out, int n) {
  int wave = (blockIdx.x * blockDim.x + threadIdx.x) >> 5;
  int lane = threadIdx.x & 31;
  const int tilesN = OUT_DIM / 16;             // 2
  int total = (n / 16) * tilesN;
  if (wave >= total) return;
  int tm = (wave / tilesN) * 16;
  int tn = (wave % tilesN) * 16;
  int l15 = lane & 15;
  int kg  = lane >> 4;

  const float*    arow = combined + (size_t)(tm + l15) * HID;
  const _Float16* brow = Woh + (size_t)(tn + l15) * HID;

  v8f acc = {};
  for (int k0 = 0; k0 < HID; k0 += 32) {
    __builtin_prefetch(arow + k0 + 32, 0, 0);
    const float* pa0 = arow + k0 + kg * 16;
    const float* pa1 = arow + k0 + kg * 16 + 8;
    v16h av;
    for (int i = 0; i < 8; i++) {
      av[i]     = (_Float16)pa0[i];
      av[i + 8] = (_Float16)pa1[i];
    }
    v16h bv = *(const v16h*)(brow + k0 + kg * 16);
    acc = __builtin_amdgcn_wmma_f32_16x16x32_f16(false, av, false, bv,
                                                 (short)0, acc, false, false);
  }
  for (int v = 0; v < 8; v++) {
    int r = tm + v + 8 * kg;
    int c = tn + l15;
    out[(size_t)r * OUT_DIM + c] = fmaxf(acc[v] + bo[c], 0.0f);
  }
}

// ---------------------------------------------------------------------------
// host-side orchestration
// ---------------------------------------------------------------------------
extern "C" void kernel_launch(void* const* d_in, const int* in_sizes, int n_in,
                              void* d_out, int out_size, void* d_ws, size_t ws_size,
                              hipStream_t stream) {
  const float* feat      = (const float*)d_in[0];
  const float* edge_pred = (const float*)d_in[1];
  const int*   src       = (const int*)d_in[2];
  const int*   dst       = (const int*)d_in[3];
  const float* Wc[4]  = {(const float*)d_in[4],  (const float*)d_in[8],
                         (const float*)d_in[12], (const float*)d_in[16]};
  const float* alc[4] = {(const float*)d_in[5],  (const float*)d_in[9],
                         (const float*)d_in[13], (const float*)d_in[17]};
  const float* arc[4] = {(const float*)d_in[6],  (const float*)d_in[10],
                         (const float*)d_in[14], (const float*)d_in[18]};
  const float* bc[4]  = {(const float*)d_in[7],  (const float*)d_in[11],
                         (const float*)d_in[15], (const float*)d_in[19]};
  const float* Wo = (const float*)d_in[20];
  const float* bo = (const float*)d_in[21];
  const float* iw = (const float*)d_in[22];

  int n = in_sizes[0] / IN_DIM;   // 100000 (multiple of 16)
  int e = in_sizes[1];            // 400000

  // workspace carve-out (256B aligned)
  char*  ws  = (char*)d_ws;
  size_t off = 0;
  auto alloc = [&](size_t bytes) -> void* {
    void* p = ws + off;
    off = (off + bytes + 255) & ~(size_t)255;
    return p;
  };
  _Float16* featf16 = (_Float16*)alloc((size_t)n * IN_DIM * 2);
  float*    hbuf    = (float*)   alloc((size_t)n * HID * 4);
  float*    combined= (float*)   alloc((size_t)n * HID * 4);
  float*    el      = (float*)   alloc((size_t)n * HEADS * 4);
  float*    er      = (float*)   alloc((size_t)n * HEADS * 4);
  unsigned* segmax  = (unsigned*)alloc((size_t)n * HEADS * 4);
  float*    segsum  = (float*)   alloc((size_t)n * HEADS * 4);
  float*    w4      = (float*)   alloc((size_t)4 * e * 4);
  _Float16* Wf16    = (_Float16*)alloc((size_t)HID * IN_DIM * 2);
  _Float16* Wof16   = (_Float16*)alloc((size_t)OUT_DIM * HID * 2);
  float*    thr     = (float*)   alloc(2 * 4);
  float*    scales  = (float*)   alloc(4 * 4);
  (void)ws_size; (void)n_in; (void)out_size;

  // 1) normalize + f16 copy (wave per row)
  k_norm_feat<<<(n + 7) / 8, 256, 0, stream>>>(feat, featf16, n);

  // 2) thresholds: neg quantile q=0.2 over p<=0, pos quantile q=0.8 over p>0
  k_quantile<<<1, 1024, 0, stream>>>(edge_pred, e, 0, 0.2f, thr + 0);
  k_quantile<<<1, 1024, 0, stream>>>(edge_pred, e, 1, 0.8f, thr + 1);

  // 3) combination scales
  k_scales<<<1, 1, 0, stream>>>(iw, scales);

  // 4) the four edge-weight softmaxes (order: all, neg, pos, unk)
  k_edge_weights<<<1, 1024, 0, stream>>>(edge_pred, thr, w4, e);

  // 5) combined accumulator = scaled biases
  {
    size_t tot = (size_t)n * HID;
    k_init_combined<<<(unsigned)((tot + 255) / 256), 256, 0, stream>>>(
        bc[0], bc[1], bc[2], bc[3], scales, combined, n);
  }

  // 6-12) the four GAT convs: g(all), n(neg), p(pos), u(unk)
  int gemmBlocks = n / 16;   // one block per 16-row stripe
  for (int c = 0; c < 4; c++) {
    k_cvt_f16<<<(HID * IN_DIM + 255) / 256, 256, 0, stream>>>(Wc[c], Wf16, HID * IN_DIM);
    k_gemm_h<<<gemmBlocks, 256, 0, stream>>>(featf16, Wf16, hbuf, n);
    k_elr<<<(n * HEADS + 255) / 256, 256, 0, stream>>>(hbuf, alc[c], arc[c], el, er, n);
    k_zero_seg<<<(n * HEADS + 255) / 256, 256, 0, stream>>>(segmax, segsum, n * HEADS);
    k_edge_max<<<(e * HEADS + 255) / 256, 256, 0, stream>>>(edge_pred, thr, src, dst,
                                                            el, er, segmax, e, c);
    k_edge_sum<<<(e * HEADS + 255) / 256, 256, 0, stream>>>(edge_pred, thr, src, dst,
                                                            el, er, segmax, segsum, e, c);
    k_edge_aggr<<<(e * 32 + 255) / 256, 256, 0, stream>>>(edge_pred, thr, src, dst,
                                                          el, er, segmax, segsum,
                                                          w4 + (size_t)c * e, scales, c,
                                                          hbuf, combined, e, c);
  }

  // 13) final projection + ReLU
  k_cvt_f16<<<(OUT_DIM * HID + 255) / 256, 256, 0, stream>>>(Wo, Wof16, OUT_DIM * HID);
  int outWaves = (n / 16) * (OUT_DIM / 16);
  int outBlocks = (outWaves * 32 + 255) / 256;
  k_gemm_out<<<outBlocks, 256, 0, stream>>>(combined, Wof16, bo, (float*)d_out, n);
}
